// PostProcessSeginw_13408887898405
// MI455X (gfx1250) — matrix operand order repaired and checked by reading, not compiled, not run on hardware
//
#include <hip/hip_runtime.h>
#include <stdint.h>

// ---------------- problem constants ----------------
#define BB   32
#define QQ   900
#define TT   1024
#define CC   365
#define CPAD 368
#define NSEL 300
#define MTOT (BB * QQ)            // 28800 rows of A
#define NPQ  (QQ * CC)            // 328500 candidates per batch

// workspace layout (bytes)
#define WS_ABF  0ull                                   // 28800*1024 bf16 = 58,982,400
#define WS_BBF  58982400ull                            // 368*1024 bf16  =    753,664
#define WS_PROB 59736064ull                            // 28800*365 f32  = 42,048,000

typedef __attribute__((ext_vector_type(16))) __bf16 v16bf;
typedef __attribute__((ext_vector_type(8)))  float  v8f;
typedef __attribute__((ext_vector_type(4)))  unsigned int u32x4;
typedef __attribute__((ext_vector_type(8)))  int    i32x8;
typedef __attribute__((ext_vector_type(4)))  int    i32x4;

// ---------------- helpers ----------------
__device__ __forceinline__ unsigned short f2bf(float f) {
  unsigned u = __float_as_uint(f);
  unsigned r = (u + 0x7FFFu + ((u >> 16) & 1u)) >> 16;   // RNE
  return (unsigned short)r;
}
__device__ __forceinline__ unsigned fkey(float f) {      // order-preserving map
  unsigned u = __float_as_uint(f);
  return (u & 0x80000000u) ? ~u : (u | 0x80000000u);
}
__device__ __forceinline__ float keyf(unsigned k) {
  unsigned u = (k & 0x80000000u) ? (k & 0x7FFFFFFFu) : ~k;
  return __uint_as_float(u);
}

// ---------------- stage 1a: sigmoid(logits) -> bf16 ----------------
__global__ __launch_bounds__(256)
void sigmoid_bf16_kernel(const float* __restrict__ x, unsigned int* __restrict__ y) {
  int i = blockIdx.x * 256 + threadIdx.x;                // one thread = 2 elems
  float2 v = ((const float2*)x)[i];
  float s0 = 1.0f / (1.0f + __expf(-v.x));
  float s1 = 1.0f / (1.0f + __expf(-v.y));
  y[i] = (unsigned)f2bf(s0) | ((unsigned)f2bf(s1) << 16);
}

// ---------------- stage 1b: positive_map -> bf16, zero-pad to 368 rows ----
__global__ __launch_bounds__(256)
void posmap_bf16_kernel(const float* __restrict__ pm, unsigned short* __restrict__ out) {
  int i = blockIdx.x * 256 + threadIdx.x;                // < 368*1024
  int c = i >> 10, t = i & 1023;
  float v = (c < CC) ? pm[c * TT + t] : 0.0f;
  out[i] = f2bf(v);
}

// ---------------- stage 2: WMMA GEMM prob = Abf * Bbf^T -------------------
// block: 256 threads = 8 waves; each wave a 16x16 output tile, full K=1024.
// B tile (16 cols x 1024 K, bf16) is TDM-loaded into LDS with 16B padding per
// 1024B -> row stride 2080B.
__global__ __launch_bounds__(256)
void gemm_prob_kernel(const unsigned short* __restrict__ Abf,
                      const unsigned short* __restrict__ Bbf,
                      float* __restrict__ prob) {
  __shared__ __align__(16) unsigned char smemB[16 * 2080];

  const int tid  = threadIdx.x;
  const int lane = tid & 31;
  const int wave = tid >> 5;
  const int c0   = blockIdx.y * 16;

  if (wave == 0) {
    // Build Tensor DMA Descriptor (D#): 2D load of 16 rows x 2048B, pad 16B/1024B
    unsigned long long ga = (unsigned long long)(uintptr_t)(Bbf + (size_t)c0 * TT);
    unsigned ldsa = (unsigned)(uintptr_t)(&smemB[0]);
    u32x4 g0; i32x8 g1; i32x4 gz; i32x8 gz8;
    g0.x = 1u;                                           // count=1, user descriptor
    g0.y = ldsa;                                         // lds_addr
    g0.z = (unsigned)(ga & 0xFFFFFFFFull);               // global_addr[31:0]
    g0.w = (unsigned)((ga >> 32) & 0x01FFFFFFull) | (2u << 30);  // addr[56:32] | type=2
    g1[0] = (int)((1u << 16)        // data_size = 2 bytes
                | (1u << 20)        // pad_enable
                | (7u << 22)        // pad_interval = 256 dwords (1024B)
                | (3u << 25));      // pad_amount   = 4 dwords (16B)
    g1[1] = (int)(((unsigned)TT & 0xFFFFu) << 16);       // tensor_dim0[15:0] @ bits63:48
    g1[2] = (int)(((unsigned)CPAD & 0xFFFFu) << 16);     // tensor_dim1[15:0] @ bits95:80
    g1[3] = (int)(((unsigned)TT & 0xFFFFu) << 16);       // tile_dim0 @ bits127:112
    g1[4] = 16;                                          // tile_dim1 = 16 rows
    g1[5] = TT;                                          // tensor_dim0_stride[31:0]
    g1[6] = 0;
    g1[7] = 0;
    gz[0] = 0; gz[1] = 0; gz[2] = 0; gz[3] = 0;
    gz8[0] = 0; gz8[1] = 0; gz8[2] = 0; gz8[3] = 0;
    gz8[4] = 0; gz8[5] = 0; gz8[6] = 0; gz8[7] = 0;
    __builtin_amdgcn_tensor_load_to_lds(g0, g1, gz, gz, gz8, 0);
    __builtin_amdgcn_s_wait_tensorcnt(0);
  }
  __syncthreads();

  const int m0 = blockIdx.x * 128 + wave * 16;
  const int r  = lane & 15;
  const int kh = lane >> 4;     // 0: lanes 0-15, 1: lanes 16-31

  // A fragment addressing per ISA 16-bit A 16x32 layout:
  // lane<16 holds K = k0+0..7 and k0+16..23 ; lane>=16 holds +8 of each.
  const unsigned short* Arow = Abf + (size_t)(m0 + r) * TT + kh * 8;

  v8f acc = {0.f, 0.f, 0.f, 0.f, 0.f, 0.f, 0.f, 0.f};

  for (int k0 = 0; k0 < TT; k0 += 32) {
    uint4 a0 = *(const uint4*)(Arow + k0);
    uint4 a1 = *(const uint4*)(Arow + k0 + 16);

    // B fragment: column (c0+r), K = k0 + kh*8 .. +7 and k0 + 16 + kh*8 .. +7 (LDS, padded rows)
    int kb  = k0 + kh * 8;
    int kb2 = kb + 16;
    unsigned off0 = (unsigned)(r * 2080 + kb * 2  + (kb  >= 512 ? 16 : 0));
    unsigned off1 = (unsigned)(r * 2080 + kb2 * 2 + (kb2 >= 512 ? 16 : 0));
    uint4 b0 = *(const uint4*)(smemB + off0);
    uint4 b1 = *(const uint4*)(smemB + off1);

    union { uint4 u[2]; v16bf v; } ua, ub;
    ua.u[0] = a0; ua.u[1] = a1;
    ub.u[0] = b0; ub.u[1] = b1;
    acc = __builtin_amdgcn_wmma_f32_16x16x32_bf16(
        false, ua.v, false, ub.v, (short)0, acc, false, false);
  }

  // C/D layout: vgpr i, lanes 0-15 -> M=m0+i, lanes 16-31 -> M=m0+8+i; N=lane&15
  const int col = c0 + r;
  if (col < CC) {
    const int rb = m0 + (kh << 3);
    #pragma unroll
    for (int i2 = 0; i2 < 8; ++i2)
      prob[(size_t)(rb + i2) * CC + col] = acc[i2];
  }
}

// ---------------- stage 3: per-batch exact top-300 + box decode -----------
__global__ __launch_bounds__(1024)
void select_kernel(const float* __restrict__ prob,
                   const float* __restrict__ boxes_in,
                   const float* __restrict__ tsizes,
                   float* __restrict__ out) {
  const int b   = blockIdx.x;
  const int tid = threadIdx.x;
  const float* pb = prob + (size_t)b * NPQ;

  __shared__ unsigned hist[256];
  __shared__ unsigned sh_prefix, sh_need, cnt_gt, cnt_eq;
  __shared__ unsigned long long cand[512];

  if (tid == 0) { sh_prefix = 0u; sh_need = NSEL; }

  // 4-pass radix select: exact 300th-largest 32-bit key
  for (int pass = 0; pass < 4; ++pass) {
    const int shift = 24 - 8 * pass;
    if (tid < 256) hist[tid] = 0u;
    __syncthreads();
    const unsigned pref = sh_prefix;
    for (int i = tid; i < NPQ; i += 1024) {
      unsigned key = fkey(pb[i]);
      bool match = (pass == 0) || ((key >> (shift + 8)) == (pref >> (shift + 8)));
      if (match) atomicAdd(&hist[(key >> shift) & 255u], 1u);
    }
    __syncthreads();
    if (tid == 0) {
      unsigned need = sh_need, acc = 0; int chosen = 0;
      for (int bin = 255; bin >= 0; --bin) {
        acc += hist[bin];
        if (acc >= need) { chosen = bin; sh_need = need - (acc - hist[bin]); break; }
      }
      sh_prefix = pref | ((unsigned)chosen << shift);
    }
    __syncthreads();
  }
  const unsigned thr    = sh_prefix;
  const unsigned needEq = sh_need;
  const unsigned nGt    = NSEL - needEq;

  if (tid == 0) { cnt_gt = 0u; cnt_eq = 0u; }
  if (tid < 512) cand[tid] = 0ull;
  __syncthreads();

  for (int i = tid; i < NPQ; i += 1024) {
    unsigned key = fkey(pb[i]);
    if (key > thr) {
      unsigned p = atomicAdd(&cnt_gt, 1u);
      cand[p] = ((unsigned long long)key << 32) | (unsigned)(0xFFFFFFFFu - (unsigned)i);
    } else if (key == thr) {
      unsigned p = atomicAdd(&cnt_eq, 1u);
      if (p < needEq)
        cand[nGt + p] = ((unsigned long long)key << 32) | (unsigned)(0xFFFFFFFFu - (unsigned)i);
    }
  }
  __syncthreads();

  // bitonic sort, descending on (key desc, index asc) composite, 512 elems
  for (int k = 2; k <= 512; k <<= 1) {
    for (int j = k >> 1; j > 0; j >>= 1) {
      if (tid < 512) {
        int ixj = tid ^ j;
        if (ixj > tid) {
          unsigned long long a = cand[tid], c2 = cand[ixj];
          bool up = (tid & k) == 0;
          if (up ? (a < c2) : (a > c2)) { cand[tid] = c2; cand[ixj] = a; }
        }
      }
      __syncthreads();
    }
  }

  if (tid < NSEL) {
    unsigned long long e = cand[tid];
    unsigned key = (unsigned)(e >> 32);
    unsigned idx = 0xFFFFFFFFu - (unsigned)(e & 0xFFFFFFFFull);
    int q = (int)(idx / (unsigned)CC);
    int c = (int)(idx % (unsigned)CC);

    out[b * NSEL + tid]                 = keyf(key);   // scores
    out[BB * NSEL + b * NSEL + tid]     = (float)c;    // labels (as float)

    const float* bx = boxes_in + ((size_t)b * QQ + q) * 4;
    float cx = bx[0], cy = bx[1], w = bx[2], h = bx[3];
    float ih = tsizes[b * 2 + 0], iw = tsizes[b * 2 + 1];
    float* bo = out + 2 * BB * NSEL + ((size_t)b * NSEL + tid) * 4;
    bo[0] = (cx - 0.5f * w) * iw;
    bo[1] = (cy - 0.5f * h) * ih;
    bo[2] = (cx + 0.5f * w) * iw;
    bo[3] = (cy + 0.5f * h) * ih;
  }
}

// ---------------- launcher ----------------
extern "C" void kernel_launch(void* const* d_in, const int* in_sizes, int n_in,
                              void* d_out, int out_size, void* d_ws, size_t ws_size,
                              hipStream_t stream) {
  (void)in_sizes; (void)n_in; (void)out_size; (void)ws_size;
  const float* logits = (const float*)d_in[0];   // 32*900*1024
  const float* boxes  = (const float*)d_in[1];   // 32*900*4
  const float* pm     = (const float*)d_in[2];   // 365*1024
  const float* ts     = (const float*)d_in[3];   // 32*2

  char* ws = (char*)d_ws;
  unsigned short* Abf = (unsigned short*)(ws + WS_ABF);
  unsigned short* Bbf = (unsigned short*)(ws + WS_BBF);
  float*          prb = (float*)(ws + WS_PROB);

  sigmoid_bf16_kernel<<<(MTOT * TT) / 2 / 256, 256, 0, stream>>>(logits, (unsigned int*)Abf);
  posmap_bf16_kernel<<<(CPAD * TT) / 256, 256, 0, stream>>>(pm, Bbf);
  gemm_prob_kernel<<<dim3(MTOT / 128, CPAD / 16), 256, 0, stream>>>(Abf, Bbf, prb);
  select_kernel<<<BB, 1024, 0, stream>>>(prb, boxes, ts, (float*)d_out);
}
